// Head_34866544509086
// MI455X (gfx1250) — compile-verified, matrix operand order checked by reading, not run on hardware
//
#include <hip/hip_runtime.h>
#include <stdint.h>

// Problem dims (fixed by the reference)
#define BB 8
#define TT 2048
#define CC 1024
#define HH 64

typedef _Float16 v16h __attribute__((ext_vector_type(16)));
typedef _Float16 h8   __attribute__((ext_vector_type(8)));
typedef float    v8f  __attribute__((ext_vector_type(8)));
typedef unsigned int ui32x4 __attribute__((ext_vector_type(4)));
typedef int          i32x4  __attribute__((ext_vector_type(4)));
typedef int          i32x8  __attribute__((ext_vector_type(8)));

union V16 { v16h v; h8 h[2]; };

#if defined(__gfx1250__) && __has_builtin(__builtin_amdgcn_tensor_load_to_lds) && \
    __has_builtin(__builtin_amdgcn_s_wait_tensorcnt)
#define HAS_TDM 1
#else
#define HAS_TDM 0
#endif

__device__ __forceinline__ v8f wmma_f16(v16h a, v16h b, v8f c) {
    // D = A(16x32 f16) * B(32x16 f16) + C(16x16 f32)
    return __builtin_amdgcn_wmma_f32_16x16x32_f16(
        false, a, false, b, (short)0, c, false, false);
}

#if HAS_TDM
// Issue one TDM 2D tile load: global (row-major, stride0 elements between
// dim1 rows, f16 data) -> LDS contiguous tile [tile1][tile0].
// Descriptor bitfields per CDNA5 ISA section 8.3/8.4.
__device__ __forceinline__ void tdm_load_2d(uint32_t lds_off, const void* gptr,
                                            uint32_t dim0, uint32_t dim1,
                                            uint32_t stride0,
                                            uint32_t tile0, uint32_t tile1) {
    const uint64_t ga = (uint64_t)(uintptr_t)gptr;
    ui32x4 g0;
    g0[0] = 1u;                                        // count=1, user mode
    g0[1] = lds_off;                                   // lds_addr (bytes)
    g0[2] = (uint32_t)ga;                              // global_addr[31:0]
    g0[3] = (uint32_t)((ga >> 32) & 0x1FFFFFFu) | (2u << 30); // addr[56:32] | type=2
    i32x8 g1;
    g1[0] = (int)(1u << 16);                           // wg_mask=0, data_size=1 (2B)
    g1[1] = (int)((dim0 & 0xFFFFu) << 16);             // tensor_dim0[15:0] @ bit48
    g1[2] = (int)(((dim0 >> 16) & 0xFFFFu) | ((dim1 & 0xFFFFu) << 16));
    g1[3] = (int)(((dim1 >> 16) & 0xFFFFu) | ((tile0 & 0xFFFFu) << 16));
    g1[4] = (int)(tile1 & 0xFFFFu);                    // tile_dim1 (tile_dim2=0)
    g1[5] = (int)stride0;                              // tensor_dim0_stride[31:0]
    g1[6] = 0;                                         // stride0 hi / dim1_stride lo
    g1[7] = 0;
    i32x4 gz = {0, 0, 0, 0};
#if __clang_major__ >= 23
    i32x8 gz8 = {0, 0, 0, 0, 0, 0, 0, 0};
    __builtin_amdgcn_tensor_load_to_lds(g0, g1, gz, gz, gz8, 0);
#else
    __builtin_amdgcn_tensor_load_to_lds(g0, g1, gz, gz, 0);
#endif
}
#endif // HAS_TDM

// --------------------------------------------------------------------------
// Kernel 0: repack weights fp32[C][H] -> f16 B-fragment-linear layout so the
// projection loop reads each fragment as 2 x b128.  Layout:
//   wpack[((z*32 + kt)*4 + nt)*32 + lane][16]
// --------------------------------------------------------------------------
__global__ __launch_bounds__(256)
void pack_w_kernel(const float* __restrict__ Wq, const float* __restrict__ Wk,
                   const float* __restrict__ Wv, _Float16* __restrict__ wpack) {
    const int lane = threadIdx.x & 31;
    const int gw   = blockIdx.x * 8 + (threadIdx.x >> 5); // 0..383
    const int r = lane & 15, hi = lane >> 4;
    const int z   = gw / 128;
    const int rem = gw % 128;
    const int kt  = rem >> 2;       // 32 K-tiles of 32
    const int nt  = rem & 3;        // 4 N-tiles of 16
    const float* W = (z == 0) ? Wq : ((z == 1) ? Wk : Wv);
    const int kk = kt * 32, n0 = nt * 16;

    V16 b;
#pragma unroll
    for (int j = 0; j < 8; ++j) {
        b.v[j]     = (_Float16)W[(size_t)(kk + hi * 8 + j)      * HH + n0 + r];
        b.v[j + 8] = (_Float16)W[(size_t)(kk + hi * 8 + 16 + j) * HH + n0 + r];
    }
    _Float16* dst = wpack + ((size_t)gw * 32 + lane) * 16;
    *reinterpret_cast<h8*>(dst)     = b.h[0];
    *reinterpret_cast<h8*>(dst + 8) = b.h[1];
}

// --------------------------------------------------------------------------
// Kernel 1: QKV projection.  Q = q@Wq, K = k@Wk, V = k@Wv  (fp32 -> f16)
// ws layout: [0,N) Qf16 [B*T][H]; [N,2N) Kf16 [B*T][H]; [2N,3N) V^T [B][H][T];
//            [3N,...) packed weights.   N = B*T*H
// --------------------------------------------------------------------------
__global__ __launch_bounds__(256)
void qkv_proj_kernel(const float* __restrict__ q, const float* __restrict__ k,
                     _Float16* __restrict__ ws16) {
    const int lane = threadIdx.x & 31;
    const int wave = threadIdx.x >> 5;
    const int r  = lane & 15;
    const int hi = lane >> 4;
    const int nt = wave & 3;
    const int mt = wave >> 2;
    const int m0 = blockIdx.x * 32 + mt * 16;
    const int n0 = nt * 16;
    const int z  = blockIdx.y;     // 0=Q, 1=K, 2=V

    const size_t N = (size_t)BB * TT * HH;
    const float* src = (z == 0) ? q : k;
    const _Float16* wp = ws16 + 3 * N + ((size_t)z * 128) * 32 * 16;

    v8f acc = {};
    const float* arow = src + (size_t)(m0 + r) * CC;

    for (int kt = 0; kt < 32; ++kt) {
        const int kk = kt * 32;
        // A fragment (16x32): row m0+r, K = {kk+hi*8+j, kk+hi*8+16+j}
        const float4 f0 = *reinterpret_cast<const float4*>(arow + kk + hi * 8);
        const float4 f1 = *reinterpret_cast<const float4*>(arow + kk + hi * 8 + 4);
        const float4 f2 = *reinterpret_cast<const float4*>(arow + kk + hi * 8 + 16);
        const float4 f3 = *reinterpret_cast<const float4*>(arow + kk + hi * 8 + 20);
        V16 a;
        a.v[0]  = (_Float16)f0.x; a.v[1]  = (_Float16)f0.y; a.v[2]  = (_Float16)f0.z; a.v[3]  = (_Float16)f0.w;
        a.v[4]  = (_Float16)f1.x; a.v[5]  = (_Float16)f1.y; a.v[6]  = (_Float16)f1.z; a.v[7]  = (_Float16)f1.w;
        a.v[8]  = (_Float16)f2.x; a.v[9]  = (_Float16)f2.y; a.v[10] = (_Float16)f2.z; a.v[11] = (_Float16)f2.w;
        a.v[12] = (_Float16)f3.x; a.v[13] = (_Float16)f3.y; a.v[14] = (_Float16)f3.z; a.v[15] = (_Float16)f3.w;

        // B fragment: 2 x b128 from packed weights
        const _Float16* bp = wp + (((size_t)kt * 4 + nt) * 32 + lane) * 16;
        V16 b;
        b.h[0] = *reinterpret_cast<const h8*>(bp);
        b.h[1] = *reinterpret_cast<const h8*>(bp + 8);

        acc = wmma_f16(a.v, b.v, acc);
    }

    if (z < 2) {
        _Float16* dst = ws16 + (size_t)z * N;      // row-major [B*T][H]
#pragma unroll
        for (int v = 0; v < 8; ++v) {
            const int row = m0 + hi * 8 + v;       // C-layout rows
            dst[(size_t)row * HH + n0 + r] = (_Float16)acc[v];
        }
    } else {
        _Float16* vt = ws16 + 2 * N;               // transposed [B][H][T]
#pragma unroll
        for (int v = 0; v < 8; ++v) {
            const int row = m0 + hi * 8 + v;
            const int b_  = row / TT, t = row % TT;
            vt[((size_t)b_ * HH + n0 + r) * TT + t] = (_Float16)acc[v];
        }
    }
}

// --------------------------------------------------------------------------
// Kernel 2: causal flash attention.  4 waves/block cover 64 consecutive query
// rows of one batch; the block streams a SHARED key-tile sequence (32 keys per
// step).  Wave 0 drives a double-buffered Tensor-Data-Mover pipeline: while
// the block computes on buf[i&1], the TDM fills buf[(i+1)&1]; s_wait_tensorcnt(2)
// exploits in-order TENSORcnt completion, __syncthreads publishes the tile.
// Per step: 8 WMMAs, online softmax via shfl_xor, LDS bounce for P re-layout.
// --------------------------------------------------------------------------
__global__ __launch_bounds__(128)
void attn_kernel(const _Float16* __restrict__ ws16, float* __restrict__ out) {
#if HAS_TDM
    __shared__ _Float16 Kbuf[2][32][64];           // 8 KB: double-buffered K tile
    __shared__ _Float16 Vbuf[2][64][32];           // 8 KB: double-buffered V tile
#endif
    __shared__ _Float16 Plds[4][16][32];           // 4 KB: P transpose bounce

    const int lane = threadIdx.x & 31;
    const int wave = threadIdx.x >> 5;
    const int r  = lane & 15;
    const int hi = lane >> 4;
    const int g0r = blockIdx.x * 64 + wave * 16;   // global query row tile start
    const int b   = g0r / TT;
    const int t0  = g0r % TT;                      // this wave's query start
    const int t0blk = (blockIdx.x * 64) % TT;      // block's query start
    const int ntiles = t0blk / 32 + 2;             // 32-key tiles: covers t0blk+63

    const size_t N = (size_t)BB * TT * HH;
    const _Float16* Q  = ws16;
    const _Float16* Kb = ws16 + N     + (size_t)b * TT * HH;   // [T][H]
    const _Float16* Vb = ws16 + 2 * N + (size_t)b * HH * TT;   // [H][T]

    // Q A-fragments (16x64 tile = two K-steps of 32)
    const _Float16* qrow = Q + (size_t)(g0r + r) * HH;
    V16 a0, a1;
    a0.h[0] = *reinterpret_cast<const h8*>(qrow + hi * 8);
    a0.h[1] = *reinterpret_cast<const h8*>(qrow + hi * 8 + 16);
    a1.h[0] = *reinterpret_cast<const h8*>(qrow + 32 + hi * 8);
    a1.h[1] = *reinterpret_cast<const h8*>(qrow + 32 + hi * 8 + 16);

    float m8[8], l8[8];
    v8f o0 = {}, o1 = {}, o2 = {}, o3 = {};
#pragma unroll
    for (int v = 0; v < 8; ++v) { m8[v] = -__builtin_inff(); l8[v] = 0.0f; }

    const float scale = 0.03125f;                  // C^-0.5 = 1/32

#if HAS_TDM
    // Prologue: DMA tile 0 into buffer 0 (one wave drives the TDM).
    if (wave == 0) {
        tdm_load_2d((uint32_t)(uintptr_t)&Kbuf[0][0][0], Kb,
                    HH, 32, HH, HH, 32);
        tdm_load_2d((uint32_t)(uintptr_t)&Vbuf[0][0][0], Vb,
                    32, HH, TT, 32, HH);
    }
#endif

    for (int it = 0; it < ntiles; ++it) {
        const int s0  = it * 32;
        const int cur = it & 1;

#if HAS_TDM
        if (wave == 0) {
            if (it + 1 < ntiles) {
                const int sn  = s0 + 32;
                const int nxt = cur ^ 1;
                tdm_load_2d((uint32_t)(uintptr_t)&Kbuf[nxt][0][0],
                            Kb + (size_t)sn * HH, HH, 32, HH, HH, 32);
                tdm_load_2d((uint32_t)(uintptr_t)&Vbuf[nxt][0][0],
                            Vb + sn, 32, HH, TT, 32, HH);
                asm volatile("" ::: "memory");
                __builtin_amdgcn_s_wait_tensorcnt((short)2); // current tile done
            } else {
                asm volatile("" ::: "memory");
                __builtin_amdgcn_s_wait_tensorcnt((short)0);
            }
            asm volatile("" ::: "memory");
        }
        __syncthreads();                            // publish buf[cur] to block
#endif

        if (s0 <= t0 + 15) {                       // wave-uniform: skip masked tiles
#if HAS_TDM
            const _Float16* k0 = &Kbuf[cur][r][0];
            const _Float16* k1 = &Kbuf[cur][16 + r][0];
#else
            const _Float16* k0 = Kb + (size_t)(s0 + r) * HH;
            const _Float16* k1 = Kb + (size_t)(s0 + 16 + r) * HH;
#endif
            // ---- K B-fragments: two 16-key tiles x two h-steps ----------
            V16 b00, b01, b10, b11;
            b00.h[0] = *reinterpret_cast<const h8*>(k0 + hi * 8);
            b00.h[1] = *reinterpret_cast<const h8*>(k0 + hi * 8 + 16);
            b01.h[0] = *reinterpret_cast<const h8*>(k0 + 32 + hi * 8);
            b01.h[1] = *reinterpret_cast<const h8*>(k0 + 32 + hi * 8 + 16);
            b10.h[0] = *reinterpret_cast<const h8*>(k1 + hi * 8);
            b10.h[1] = *reinterpret_cast<const h8*>(k1 + hi * 8 + 16);
            b11.h[0] = *reinterpret_cast<const h8*>(k1 + 32 + hi * 8);
            b11.h[1] = *reinterpret_cast<const h8*>(k1 + 32 + hi * 8 + 16);

            v8f zc = {};
            v8f slo = wmma_f16(a0.v, b00.v, zc);
            slo     = wmma_f16(a1.v, b01.v, slo);
            v8f shi = wmma_f16(a0.v, b10.v, zc);
            shi     = wmma_f16(a1.v, b11.v, shi);

            // ---- causal mask + scale + online softmax -------------------
#pragma unroll
            for (int v = 0; v < 8; ++v) {
                const int qrg = t0 + hi * 8 + v;
                float sl = slo[v] * scale;
                float sh = shi[v] * scale;
                if (s0 + r      > qrg) sl = -__builtin_inff();
                if (s0 + 16 + r > qrg) sh = -__builtin_inff();

                float rm = fmaxf(sl, sh);
#pragma unroll
                for (int mx = 1; mx < 16; mx <<= 1)
                    rm = fmaxf(rm, __shfl_xor(rm, mx, 32));

                const float newm  = fmaxf(m8[v], rm);
                const float alpha = __expf(m8[v] - newm);
                m8[v] = newm;
                const float pl = __expf(sl - newm);
                const float ph = __expf(sh - newm);
                float ps = pl + ph;
#pragma unroll
                for (int mx = 1; mx < 16; mx <<= 1)
                    ps += __shfl_xor(ps, mx, 32);
                l8[v] = l8[v] * alpha + ps;

                o0[v] *= alpha; o1[v] *= alpha; o2[v] *= alpha; o3[v] *= alpha;

                Plds[wave][hi * 8 + v][r]      = (_Float16)pl;
                Plds[wave][hi * 8 + v][16 + r] = (_Float16)ph;
            }

            __builtin_amdgcn_wave_barrier();

            // ---- reload P as A-fragment (16x32) -------------------------
            V16 pa;
            pa.h[0] = *reinterpret_cast<const h8*>(&Plds[wave][r][hi * 8]);
            pa.h[1] = *reinterpret_cast<const h8*>(&Plds[wave][r][hi * 8 + 16]);
            __builtin_amdgcn_wave_barrier();

            // ---- O += P @ V  (4 n-tiles) --------------------------------
#if HAS_TDM
#define VROW(n) (&Vbuf[cur][(n) * 16 + r][0])
#else
#define VROW(n) (Vb + (size_t)((n) * 16 + r) * TT + s0)
#endif
            {
                V16 vb;
                vb.h[0] = *reinterpret_cast<const h8*>(VROW(0) + hi * 8);
                vb.h[1] = *reinterpret_cast<const h8*>(VROW(0) + hi * 8 + 16);
                o0 = wmma_f16(pa.v, vb.v, o0);
            }
            {
                V16 vb;
                vb.h[0] = *reinterpret_cast<const h8*>(VROW(1) + hi * 8);
                vb.h[1] = *reinterpret_cast<const h8*>(VROW(1) + hi * 8 + 16);
                o1 = wmma_f16(pa.v, vb.v, o1);
            }
            {
                V16 vb;
                vb.h[0] = *reinterpret_cast<const h8*>(VROW(2) + hi * 8);
                vb.h[1] = *reinterpret_cast<const h8*>(VROW(2) + hi * 8 + 16);
                o2 = wmma_f16(pa.v, vb.v, o2);
            }
            {
                V16 vb;
                vb.h[0] = *reinterpret_cast<const h8*>(VROW(3) + hi * 8);
                vb.h[1] = *reinterpret_cast<const h8*>(VROW(3) + hi * 8 + 16);
                o3 = wmma_f16(pa.v, vb.v, o3);
            }
#undef VROW
        }

#if HAS_TDM
        __syncthreads();       // all readers done before buf[cur] is re-filled
#endif
    }

    // ---- epilogue: normalize by l and store fp32 ------------------------
#pragma unroll
    for (int v = 0; v < 8; ++v) {
        const float inv = 1.0f / l8[v];
        float* dst = out + (size_t)(g0r + hi * 8 + v) * HH;
        dst[ 0 + r] = o0[v] * inv;
        dst[16 + r] = o1[v] * inv;
        dst[32 + r] = o2[v] * inv;
        dst[48 + r] = o3[v] * inv;
    }
}

// --------------------------------------------------------------------------
extern "C" void kernel_launch(void* const* d_in, const int* in_sizes, int n_in,
                              void* d_out, int out_size, void* d_ws, size_t ws_size,
                              hipStream_t stream) {
    (void)in_sizes; (void)n_in; (void)out_size; (void)ws_size;
    const float* q  = (const float*)d_in[0];
    const float* k  = (const float*)d_in[1];
    const float* Wq = (const float*)d_in[2];
    const float* Wk = (const float*)d_in[3];
    const float* Wv = (const float*)d_in[4];
    _Float16* ws16 = (_Float16*)d_ws;   // 3*B*T*H f16 + 384KB packed W  (~7 MB)
    float* out = (float*)d_out;

    const size_t N = (size_t)BB * TT * HH;
    _Float16* wpack = ws16 + 3 * N;

    pack_w_kernel<<<48, 256, 0, stream>>>(Wq, Wk, Wv, wpack);

    dim3 gProj(BB * TT / 32, 3);
    qkv_proj_kernel<<<gProj, 256, 0, stream>>>(q, k, ws16);

    attn_kernel<<<BB * TT / 64, 128, 0, stream>>>(ws16, out);
}
